// SparseAttention_36764920054376
// MI455X (gfx1250) — matrix operand order repaired and checked by reading, not compiled, not run on hardware
//
#include <hip/hip_runtime.h>
#include <hip/hip_bf16.h>
#include <math.h>

// ---------------- problem constants ----------------
#define BB   2
#define NN   4096
#define DIMD 2048
#define HH   16
#define DHD  128
#define BSS  32
#define SELS 16
#define WW   128          // N/BS
#define HID  4096         // BS*DH
#define BH   (BB*HH)      // 32
#define MCO  129          // MEM + W compressed slots
#define FM   (SELS*BSS)   // 512 fine keys
#define SCALEF 0.08838834764831845f   // DH^-0.5
#define EPSF   1.1920929e-07f
#define NEGF  (-1e30f)

// ---------------- WMMA fragment types ----------------
typedef __attribute__((ext_vector_type(16))) __bf16        bf16x16;
typedef __attribute__((ext_vector_type(8)))  float         f32x8;
typedef __attribute__((ext_vector_type(4)))  unsigned int  u32x4;

union FragAB { bf16x16 v; unsigned short u[16]; u32x4 q[2]; };
union FragC  { f32x8 v; float f[8]; };

__device__ __forceinline__ unsigned short f2bf(float f) {
  unsigned int u = __builtin_bit_cast(unsigned int, f);
  u += 0x7FFFu + ((u >> 16) & 1u);          // round-to-nearest-even
  return (unsigned short)(u >> 16);
}

__device__ __forceinline__ f32x8 wmma_bf16(FragAB a, FragAB b, f32x8 c) {
  return __builtin_amdgcn_wmma_f32_16x16x32_bf16(false, a.v, false, b.v,
                                                 (short)0, c, false, false);
}

// ============================================================
// Generic bf16 GEMM:  C(MxN) = A(MxK,bf16) * Bt(NxK,bf16)
// EPI: 0=f32 store, 1=bias+relu->bf16, 2=bias->bf16, 3=f32+bf16 dual
// OL : 0=row-major MxN, 1=(B,H,N,DH) head layout (M=B*N, N=H*DH)
// Block: 256 thr = 8 waves (4M x 2N). Wave tile: 16M x 64N
// (one A fragment reused across 4 WMMAs per K-step).
// Block tile: 64M x 128N.
// ============================================================
template<int EPI, int OL>
__global__ __launch_bounds__(256)
void gemm_bf16_kernel(const unsigned short* __restrict__ A,
                      const unsigned short* __restrict__ Bt,
                      const float* __restrict__ bias,
                      float* __restrict__ Cf,
                      unsigned short* __restrict__ Cb,
                      int M, int N, int K)
{
  const int lane = threadIdx.x & 31;
  const int wave = threadIdx.x >> 5;
  const int l15  = lane & 15;
  const int lhi  = lane >> 4;
  const int m0 = blockIdx.y * 64  + (wave >> 1) * 16;   // wave's 16 M-rows
  const int n0 = blockIdx.x * 128 + (wave & 1) * 64;    // wave's 64 N-cols

  FragC acc[4];
#pragma unroll
  for (int t = 0; t < 4; t++)
#pragma unroll
    for (int r = 0; r < 8; r++) acc[t].f[r] = 0.f;

  const unsigned short* arow  = A  + (size_t)(m0 + l15) * K + lhi * 8;
  const unsigned short* brow0 = Bt + (size_t)(n0 + l15) * K + lhi * 8;
  const size_t bstride = (size_t)16 * K;                // between n-tiles

  for (int k0 = 0; k0 < K; k0 += 32) {
    FragAB a;
    a.q[0] = *(const u32x4*)(arow + k0);
    a.q[1] = *(const u32x4*)(arow + k0 + 16);
    // CDNA5 prefetch (global_prefetch_b8) 8 K-steps (256 elements) ahead on both streams
    __builtin_prefetch(arow  + k0 + 256, 0, 1);
    __builtin_prefetch(brow0 + k0 + 256, 0, 1);
#pragma unroll
    for (int t = 0; t < 4; t++) {
      FragAB b;
      const unsigned short* br = brow0 + (size_t)t * bstride + k0;
      b.q[0] = *(const u32x4*)(br);
      b.q[1] = *(const u32x4*)(br + 16);
      acc[t].v = wmma_bf16(a, b, acc[t].v);
    }
  }

#pragma unroll
  for (int t = 0; t < 4; t++) {
    const int ncol = n0 + t * 16 + l15;
    float bv = 0.f;
    if constexpr (EPI == 1 || EPI == 2) bv = bias[ncol];
#pragma unroll
    for (int r = 0; r < 8; r++) {
      int mrow = m0 + r + lhi * 8;
      float vv = acc[t].f[r];
      if constexpr (EPI == 1) vv = fmaxf(vv + bv, 0.f);
      if constexpr (EPI == 2) vv = vv + bv;
      size_t oidx;
      if constexpr (OL == 0) {
        oidx = (size_t)mrow * N + ncol;
      } else {
        int bb = mrow >> 12, nn = mrow & 4095;
        int hh = ncol >> 7,  dd = ncol & 127;
        oidx = (((size_t)bb * HH + hh) * NN + nn) * DHD + dd;
      }
      if constexpr (EPI == 0)      { Cf[oidx] = vv; }
      else if constexpr (EPI == 3) { Cf[oidx] = vv; Cb[oidx] = f2bf(vv); }
      else                         { Cb[oidx] = f2bf(vv); }
    }
  }
}

// ---------------- transpose-cast:  Wt[n*K+k] = bf16(W[k*N+n]) ----------------
__global__ void tcast_kernel(const float* __restrict__ W,
                             unsigned short* __restrict__ Wt, int K, int N)
{
  size_t g = (size_t)blockIdx.x * 256 + threadIdx.x;
  size_t total = (size_t)K * N;
  if (g >= total) return;
  int n = (int)(g % N);
  int k = (int)(g / N);
  Wt[(size_t)n * K + k] = f2bf(W[g]);
}

// ---------------- elementwise cast f32 -> bf16 ----------------
__global__ void ecast_kernel(const float* __restrict__ X,
                             unsigned short* __restrict__ Y, size_t total)
{
  size_t g = (size_t)blockIdx.x * 256 + threadIdx.x;
  if (g >= total) return;
  Y[g] = f2bf(X[g]);
}

// ---------------- kc/vc build:  kc[r][i] = bf16(k[r*4096+i] + pos[h][i]) -----
__global__ void build_kc_kernel(const float* __restrict__ kf,
                                const float* __restrict__ pos,
                                unsigned short* __restrict__ kc)
{
  size_t g = (size_t)blockIdx.x * 256 + threadIdx.x;
  size_t total = (size_t)BH * WW * HID;     // 16.78M
  if (g >= total) return;
  int    i = (int)(g & (HID - 1));
  size_t r = g >> 12;                        // row over (b,h,w)
  int    h = (int)((r >> 7) & (HH - 1));     // W = 128
  kc[g] = f2bf(kf[g] + pos[(size_t)h * HID + i]);
}

// ---------------- assemble ck/cv (prepend mem slot) ----------------
__global__ void assemble_ckv_kernel(const unsigned short* __restrict__ cktmp,
                                    const unsigned short* __restrict__ cvtmp,
                                    const float* __restrict__ mem_kv,
                                    unsigned short* __restrict__ ckf,
                                    unsigned short* __restrict__ cvf)
{
  size_t g = (size_t)blockIdx.x * 256 + threadIdx.x;
  size_t total = (size_t)BH * MCO * DHD;
  if (g >= total) return;
  int    d  = (int)(g & (DHD - 1));
  size_t t  = g >> 7;
  int    m  = (int)(t % MCO);
  size_t bh = t / MCO;
  int    h  = (int)(bh & (HH - 1));
  if (m == 0) {
    ckf[g] = f2bf(mem_kv[(size_t)0 * HH * DHD + h * DHD + d]);
    cvf[g] = f2bf(mem_kv[(size_t)1 * HH * DHD + h * DHD + d]);
  } else {
    size_t src = (bh * WW + (m - 1)) * DHD + d;
    ckf[g] = cktmp[src];
    cvf[g] = cvtmp[src];
  }
}

// ---------------- gates: rmsnorm(inp,g) @ Wg + bg -> sigmoid ----------------
__global__ __launch_bounds__(256)
void gates_kernel(const float* __restrict__ inp, const float* __restrict__ gvec,
                  const float* __restrict__ Wg,  const float* __restrict__ bg,
                  float* __restrict__ gates)
{
  __shared__ float red[256];
  __shared__ float accs[32];
  const int row = blockIdx.x;
  const int tid = threadIdx.x;
  const float* x = inp + (size_t)row * DIMD;

  float ss = 0.f;
  for (int c = tid; c < DIMD; c += 256) { float v = x[c]; ss += v * v; }
  red[tid] = ss; __syncthreads();
  for (int off = 128; off > 0; off >>= 1) {
    if (tid < off) red[tid] += red[tid + off];
    __syncthreads();
  }
  const float rms = rsqrtf(red[0] * (1.f / DIMD) + EPSF);

  if (tid < 32) accs[tid] = 0.f;
  __syncthreads();

  float acc[32];
#pragma unroll
  for (int j = 0; j < 32; j++) acc[j] = 0.f;
  for (int c = tid; c < DIMD; c += 256) {
    float xv = x[c] * rms * gvec[c];
    const float* wr = Wg + (size_t)c * 32;
#pragma unroll
    for (int j = 0; j < 32; j++) acc[j] += xv * wr[j];
  }
#pragma unroll
  for (int j = 0; j < 32; j++) atomicAdd(&accs[j], acc[j]);
  __syncthreads();
  if (tid < 32)
    gates[(size_t)row * 32 + tid] = 1.f / (1.f + __expf(-(accs[tid] + bg[tid])));
}

// ---------------- coarse attention: per (bh, 16 q-rows) ----------------
__global__ __launch_bounds__(32)
void coarse_attn_kernel(const unsigned short* __restrict__ qb,
                        const unsigned short* __restrict__ ckf,
                        const unsigned short* __restrict__ cvf,
                        float* __restrict__ cof, float* __restrict__ imp)
{
  __shared__ float          sim[16][144];
  __shared__ unsigned short att[16][160];
  const int lane = threadIdx.x;
  const int l15 = lane & 15, lhi = lane >> 4;
  const int bh = blockIdx.y;
  const int q0 = blockIdx.x * 16;

  const unsigned short* qrow = qb  + ((size_t)bh * NN + q0 + l15) * DHD + lhi * 8;
  const unsigned short* ckb  = ckf + (size_t)bh * MCO * DHD;
  const unsigned short* cvb  = cvf + (size_t)bh * MCO * DHD;

  // ---- sim = (q . ck^T) * SCALE ----
  for (int t = 0; t < 9; t++) {
    FragC acc;
#pragma unroll
    for (int r = 0; r < 8; r++) acc.f[r] = 0.f;
    const int ncol = t * 16 + l15;
    for (int k0 = 0; k0 < DHD; k0 += 32) {
      FragAB a, b;
      a.q[0] = *(const u32x4*)(qrow + k0);
      a.q[1] = *(const u32x4*)(qrow + k0 + 16);
      if (ncol < MCO) {
        const unsigned short* br = ckb + (size_t)ncol * DHD + k0 + lhi * 8;
        b.q[0] = *(const u32x4*)(br);
        b.q[1] = *(const u32x4*)(br + 16);
      } else {
#pragma unroll
        for (int e = 0; e < 16; e++) b.u[e] = 0;
      }
      acc.v = wmma_bf16(a, b, acc.v);
    }
#pragma unroll
    for (int r = 0; r < 8; r++)
      sim[r + lhi * 8][ncol] = (ncol < MCO) ? acc.f[r] * SCALEF : NEGF;
  }
  __syncthreads();

  // ---- softmax over 129 slots (lane = row) ----
  if (lane < 16) {
    float mx = -INFINITY;
    for (int c = 0; c < MCO; c++) mx = fmaxf(mx, sim[lane][c]);
    float s = 0.f;
    for (int c = 0; c < MCO; c++) { float p = __expf(sim[lane][c] - mx); sim[lane][c] = p; s += p; }
    float inv = 1.f / s;
    for (int c = 0; c < MCO; c++) { float p = sim[lane][c] * inv; sim[lane][c] = p; att[lane][c] = f2bf(p); }
    for (int c = MCO; c < 160; c++) att[lane][c] = 0;
  }
  __syncthreads();

  // ---- importance accumulation (skip mem slot 0) ----
#pragma unroll
  for (int cc = 0; cc < 4; cc++) {
    int c = 1 + lane * 4 + cc;            // 1..128
    float s = 0.f;
    for (int r = 0; r < 16; r++) s += sim[r][c];
    atomicAdd(&imp[(size_t)bh * WW + (c - 1)], s);
  }

  // ---- c_out = attn @ cv ----
  for (int t = 0; t < 8; t++) {
    FragC acc;
#pragma unroll
    for (int r = 0; r < 8; r++) acc.f[r] = 0.f;
    const int ncol = t * 16 + l15;
    for (int k0 = 0; k0 < 160; k0 += 32) {
      FragAB a, b;
      const unsigned short* ar = &att[l15][k0 + lhi * 8];
      a.q[0] = *(const u32x4*)(ar);
      a.q[1] = *(const u32x4*)(ar + 16);
      const int kb = k0 + lhi * 8;
#pragma unroll
      for (int e = 0; e < 16; e++) {
        int kr = kb + (e & 7) + ((e >> 3) << 4);
        b.u[e] = (kr < MCO) ? cvb[(size_t)kr * DHD + ncol] : (unsigned short)0;
      }
      acc.v = wmma_bf16(a, b, acc.v);
    }
#pragma unroll
    for (int r = 0; r < 8; r++)
      cof[((size_t)bh * NN + q0 + r + lhi * 8) * DHD + ncol] = acc.f[r];
  }
}

// ---------------- top-16 selection per (b,h) ----------------
__global__ void topk_kernel(const float* __restrict__ imp,
                            int* __restrict__ idxb, int* __restrict__ maskb)
{
  const int bh = blockIdx.x;
  if (threadIdx.x != 0) return;
  float v[WW];
  for (int c = 0; c < WW; c++) v[c] = imp[(size_t)bh * WW + c];
  for (int s = 0; s < SELS; s++) {
    int best = 0; float bv = v[0];
    for (int c = 1; c < WW; c++) if (v[c] > bv) { bv = v[c]; best = c; }
    idxb[bh * SELS + s]  = best;
    maskb[bh * SELS + s] = ((bv * (1.f / NN)) > 1e-10f) ? 1 : 0;
    v[best] = -INFINITY;
  }
}

// ---------------- RoPE + cast to bf16 ----------------
__global__ void rope_cast_kernel(const float* __restrict__ src,
                                 unsigned short* __restrict__ dst)
{
  size_t g = (size_t)blockIdx.x * 256 + threadIdx.x;
  size_t total = (size_t)BH * NN * (DHD / 2);
  if (g >= total) return;
  int    i   = (int)(g & 63);
  size_t row = g >> 6;                       // bh*N + n
  int    n   = (int)(row & (NN - 1));
  float freq = __expf(-9.210340372f * (float)i * (1.f / 64.f)); // 10000^(-2i/128)
  float ang  = (float)n * freq;
  float c = __cosf(ang), s = __sinf(ang);
  size_t base = row * DHD + 2 * i;
  float x1 = src[base], x2 = src[base + 1];
  dst[base]     = f2bf(x1 * c - x2 * s);
  dst[base + 1] = f2bf(x1 * s + x2 * c);
}

// ---------------- fine attention: per (bh, 16 q-rows) ----------------
__global__ __launch_bounds__(32)
void fine_attn_kernel(const unsigned short* __restrict__ qr,
                      const unsigned short* __restrict__ kr,
                      const unsigned short* __restrict__ vb,
                      const int* __restrict__ idxb,
                      const int* __restrict__ maskb,
                      float* __restrict__ fof)
{
  __shared__ float          sim[16][FM];
  __shared__ unsigned short att[16][FM];
  __shared__ int lidx[SELS];
  __shared__ int lmask[SELS];
  const int lane = threadIdx.x;
  const int l15 = lane & 15, lhi = lane >> 4;
  const int bh = blockIdx.y;
  const int q0 = blockIdx.x * 16;

  if (lane < SELS) {
    lidx[lane]  = idxb[bh * SELS + lane];
    lmask[lane] = maskb[bh * SELS + lane];
  }
  __syncthreads();

  const unsigned short* qrow = qr + ((size_t)bh * NN + q0 + l15) * DHD + lhi * 8;

  // ---- fsim = (qr . fk^T)*SCALE, masked ----
  for (int t = 0; t < 32; t++) {
    const int j = t >> 1;                     // selected-block index
    const int blk = lidx[j];
    const int srow = blk * BSS + (t & 1) * 16 + l15;
    const unsigned short* brow = kr + ((size_t)bh * NN + srow) * DHD + lhi * 8;
    FragC acc;
#pragma unroll
    for (int r = 0; r < 8; r++) acc.f[r] = 0.f;
    for (int k0 = 0; k0 < DHD; k0 += 32) {
      FragAB a, b;
      a.q[0] = *(const u32x4*)(qrow + k0);
      a.q[1] = *(const u32x4*)(qrow + k0 + 16);
      b.q[0] = *(const u32x4*)(brow + k0);
      b.q[1] = *(const u32x4*)(brow + k0 + 16);
      acc.v = wmma_bf16(a, b, acc.v);
    }
    const int col = t * 16 + l15;
    const int msk = lmask[j];
#pragma unroll
    for (int r = 0; r < 8; r++)
      sim[r + lhi * 8][col] = msk ? acc.f[r] * SCALEF : NEGF;
  }
  __syncthreads();

  // ---- softmax over 512 fine keys ----
  if (lane < 16) {
    float mx = -INFINITY;
    for (int c = 0; c < FM; c++) mx = fmaxf(mx, sim[lane][c]);
    float s = 0.f;
    for (int c = 0; c < FM; c++) { float p = __expf(sim[lane][c] - mx); sim[lane][c] = p; s += p; }
    float inv = 1.f / s;
    for (int c = 0; c < FM; c++) att[lane][c] = f2bf(sim[lane][c] * inv);
  }
  __syncthreads();

  // ---- f_out = attn @ fv ----
  for (int t = 0; t < 8; t++) {
    FragC acc;
#pragma unroll
    for (int r = 0; r < 8; r++) acc.f[r] = 0.f;
    const int ncol = t * 16 + l15;
    for (int k0 = 0; k0 < FM; k0 += 32) {
      FragAB a, b;
      const unsigned short* ar = &att[l15][k0 + lhi * 8];
      a.q[0] = *(const u32x4*)(ar);
      a.q[1] = *(const u32x4*)(ar + 16);
      const int kb = k0 + lhi * 8;
#pragma unroll
      for (int e = 0; e < 16; e++) {
        int kk = kb + (e & 7) + ((e >> 3) << 4);
        int row = lidx[kk >> 5] * BSS + (kk & 31);
        b.u[e] = vb[((size_t)bh * NN + row) * DHD + ncol];
      }
      acc.v = wmma_bf16(a, b, acc.v);
    }
#pragma unroll
    for (int r = 0; r < 8; r++)
      fof[((size_t)bh * NN + q0 + r + lhi * 8) * DHD + ncol] = acc.f[r];
  }
}

// ---------------- gated combine + transpose to (B,N,H*DH) bf16 ----------------
__global__ void combine_kernel(const float* __restrict__ cof,
                               const float* __restrict__ fof,
                               const float* __restrict__ gates,
                               unsigned short* __restrict__ y)
{
  size_t g = (size_t)blockIdx.x * 256 + threadIdx.x;
  size_t total = (size_t)BH * NN * DHD;
  if (g >= total) return;
  int    d  = (int)(g & (DHD - 1));
  size_t t  = g >> 7;
  int    n  = (int)(t & (NN - 1));
  size_t bh = t >> 12;
  int    h  = (int)(bh & (HH - 1));
  int    b  = (int)(bh >> 4);
  const float* gr = gates + ((size_t)b * NN + n) * 32;
  float o = gr[2 * h] * cof[g] + gr[2 * h + 1] * fof[g];
  y[((size_t)b * NN + n) * DIMD + h * DHD + d] = f2bf(o);
}

// ============================================================
// Host orchestration
// ============================================================
extern "C" void kernel_launch(void* const* d_in, const int* in_sizes, int n_in,
                              void* d_out, int out_size, void* d_ws, size_t ws_size,
                              hipStream_t stream)
{
  (void)in_sizes; (void)n_in; (void)out_size; (void)ws_size;

  const float* inp    = (const float*)d_in[0];
  const float* gvec   = (const float*)d_in[1];
  const float* Wq     = (const float*)d_in[2];
  const float* Wk     = (const float*)d_in[3];
  const float* Wv     = (const float*)d_in[4];
  const float* k_pos  = (const float*)d_in[5];
  const float* v_pos  = (const float*)d_in[6];
  const float* mem_kv = (const float*)d_in[7];
  const float* kW1    = (const float*)d_in[8];
  const float* kb1    = (const float*)d_in[9];
  const float* kW2    = (const float*)d_in[10];
  const float* kb2    = (const float*)d_in[11];
  const float* vW1    = (const float*)d_in[12];
  const float* vb1    = (const float*)d_in[13];
  const float* vW2    = (const float*)d_in[14];
  const float* vb2    = (const float*)d_in[15];
  const float* Wg     = (const float*)d_in[16];
  const float* bg     = (const float*)d_in[17];
  const float* Wo     = (const float*)d_in[18];
  float* out = (float*)d_out;

  // ---- bump allocator over workspace ----
  char* p = (char*)d_ws;
  auto alloc = [&](size_t bytes) -> char* {
    char* r = p;
    p += (bytes + 255) & ~(size_t)255;
    return r;
  };

  const size_t ROWS = (size_t)BB * NN;          // 8192
  const size_t QE   = (size_t)BH * NN * DHD;    // 16.78M elements

  unsigned short* inpb  = (unsigned short*)alloc(ROWS * DIMD * 2);
  unsigned short* WqT   = (unsigned short*)alloc((size_t)DIMD * DIMD * 2);
  unsigned short* WkT   = (unsigned short*)alloc((size_t)DIMD * DIMD * 2);
  unsigned short* WvT   = (unsigned short*)alloc((size_t)DIMD * DIMD * 2);
  unsigned short* WoT   = (unsigned short*)alloc((size_t)DIMD * DIMD * 2);
  unsigned short* kW1T  = (unsigned short*)alloc((size_t)HID * HID * 2);
  unsigned short* vW1T  = (unsigned short*)alloc((size_t)HID * HID * 2);
  unsigned short* kW2T  = (unsigned short*)alloc((size_t)HID * DHD * 2);
  unsigned short* vW2T  = (unsigned short*)alloc((size_t)HID * DHD * 2);
  float*          qf    = (float*)alloc(QE * 4);
  float*          kf    = (float*)alloc(QE * 4);
  float*          vf    = (float*)alloc(QE * 4);
  unsigned short* qb16  = (unsigned short*)alloc(QE * 2);
  unsigned short* vb16  = (unsigned short*)alloc(QE * 2);
  unsigned short* kc    = (unsigned short*)alloc(QE * 2);
  unsigned short* vc    = (unsigned short*)alloc(QE * 2);
  unsigned short* h1k   = (unsigned short*)alloc(QE * 2);
  unsigned short* h1v   = (unsigned short*)alloc(QE * 2);
  unsigned short* cktmp = (unsigned short*)alloc((size_t)BH * WW * DHD * 2);
  unsigned short* cvtmp = (unsigned short*)alloc((size_t)BH * WW * DHD * 2);
  unsigned short* ckf   = (unsigned short*)alloc((size_t)BH * MCO * DHD * 2);
  unsigned short* cvf   = (unsigned short*)alloc((size_t)BH * MCO * DHD * 2);
  float*          imp   = (float*)alloc((size_t)BH * WW * 4);
  int*            idxb  = (int*)alloc((size_t)BH * SELS * 4);
  int*            maskb = (int*)alloc((size_t)BH * SELS * 4);
  unsigned short* qrb   = (unsigned short*)alloc(QE * 2);
  unsigned short* krb   = (unsigned short*)alloc(QE * 2);
  float*          cof   = (float*)alloc(QE * 4);
  float*          fof   = (float*)alloc(QE * 4);
  float*          gatesb= (float*)alloc(ROWS * 32 * 4);
  unsigned short* y     = (unsigned short*)alloc(ROWS * DIMD * 2);

  auto eb = [](size_t total) { return (unsigned)((total + 255) / 256); };

  // 1. transpose-cast all weights to bf16 NxK
  tcast_kernel<<<eb((size_t)DIMD * DIMD), 256, 0, stream>>>(Wq,  WqT,  DIMD, DIMD);
  tcast_kernel<<<eb((size_t)DIMD * DIMD), 256, 0, stream>>>(Wk,  WkT,  DIMD, DIMD);
  tcast_kernel<<<eb((size_t)DIMD * DIMD), 256, 0, stream>>>(Wv,  WvT,  DIMD, DIMD);
  tcast_kernel<<<eb((size_t)DIMD * DIMD), 256, 0, stream>>>(Wo,  WoT,  DIMD, DIMD);
  tcast_kernel<<<eb((size_t)HID * HID),   256, 0, stream>>>(kW1, kW1T, HID, HID);
  tcast_kernel<<<eb((size_t)HID * HID),   256, 0, stream>>>(vW1, vW1T, HID, HID);
  tcast_kernel<<<eb((size_t)HID * DHD),   256, 0, stream>>>(kW2, kW2T, HID, DHD);
  tcast_kernel<<<eb((size_t)HID * DHD),   256, 0, stream>>>(vW2, vW2T, HID, DHD);

  // 2. cast input
  ecast_kernel<<<eb(ROWS * DIMD), 256, 0, stream>>>(inp, inpb, ROWS * DIMD);

  // 3. Q/K/V projections (head layout); Q,V dual f32+bf16 stores
  {
    dim3 grid(DIMD / 128, (unsigned)(ROWS / 64));
    gemm_bf16_kernel<3, 1><<<grid, 256, 0, stream>>>(inpb, WqT, nullptr, qf, qb16,
                                                     (int)ROWS, DIMD, DIMD);
    gemm_bf16_kernel<0, 1><<<grid, 256, 0, stream>>>(inpb, WkT, nullptr, kf, nullptr,
                                                     (int)ROWS, DIMD, DIMD);
    gemm_bf16_kernel<3, 1><<<grid, 256, 0, stream>>>(inpb, WvT, nullptr, vf, vb16,
                                                     (int)ROWS, DIMD, DIMD);
  }

  // 4. gates (rmsnorm fused)
  gates_kernel<<<(unsigned)ROWS, 256, 0, stream>>>(inp, gvec, Wg, bg, gatesb);

  // 5. build compression-MLP inputs
  build_kc_kernel<<<eb(QE), 256, 0, stream>>>(kf, k_pos, kc);
  build_kc_kernel<<<eb(QE), 256, 0, stream>>>(vf, v_pos, vc);

  // 6. compression MLPs (relu layer then projection to DH)
  {
    dim3 g1(HID / 128, (BH * WW) / 64);
    gemm_bf16_kernel<1, 0><<<g1, 256, 0, stream>>>(kc, kW1T, kb1, nullptr, h1k,
                                                   BH * WW, HID, HID);
    gemm_bf16_kernel<1, 0><<<g1, 256, 0, stream>>>(vc, vW1T, vb1, nullptr, h1v,
                                                   BH * WW, HID, HID);
    dim3 g2(1, (BH * WW) / 64);                   // N = 128 -> one block col
    gemm_bf16_kernel<2, 0><<<g2, 256, 0, stream>>>(h1k, kW2T, kb2, nullptr, cktmp,
                                                   BH * WW, DHD, HID);
    gemm_bf16_kernel<2, 0><<<g2, 256, 0, stream>>>(h1v, vW2T, vb2, nullptr, cvtmp,
                                                   BH * WW, DHD, HID);
  }

  // 7. prepend mem slot
  assemble_ckv_kernel<<<eb((size_t)BH * MCO * DHD), 256, 0, stream>>>(
      cktmp, cvtmp, mem_kv, ckf, cvf);

  // 8. coarse attention (+ importance accumulation)
  hipMemsetAsync(imp, 0, (size_t)BH * WW * sizeof(float), stream);
  {
    dim3 grid(NN / 16, BH);
    coarse_attn_kernel<<<grid, 32, 0, stream>>>(qb16, ckf, cvf, cof, imp);
  }

  // 9. top-k block selection
  topk_kernel<<<BH, 32, 0, stream>>>(imp, idxb, maskb);

  // 10. RoPE on q and k (-> bf16)
  rope_cast_kernel<<<eb((size_t)BH * NN * (DHD / 2)), 256, 0, stream>>>(qf, qrb);
  rope_cast_kernel<<<eb((size_t)BH * NN * (DHD / 2)), 256, 0, stream>>>(kf, krb);

  // 11. fine attention over gathered blocks
  {
    dim3 grid(NN / 16, BH);
    fine_attn_kernel<<<grid, 32, 0, stream>>>(qrb, krb, vb16, idxb, maskb, fof);
  }

  // 12. gated combine -> (B,N,H*DH) bf16
  combine_kernel<<<eb(QE), 256, 0, stream>>>(cof, fof, gatesb, y);

  // 13. output projection -> d_out f32
  {
    dim3 grid(DIMD / 128, (unsigned)(ROWS / 64));
    gemm_bf16_kernel<0, 0><<<grid, 256, 0, stream>>>(y, WoT, nullptr, out, nullptr,
                                                     (int)ROWS, DIMD, DIMD);
  }
}
